// VQCClassifier_23398981829253
// MI455X (gfx1250) — matrix operand order repaired and checked by reading, not compiled, not run on hardware
//
#include <hip/hip_runtime.h>
#include <hip/hip_bf16.h>
#include <math.h>

typedef __attribute__((ext_vector_type(16))) _Float16 v16h;
typedef __attribute__((ext_vector_type(8)))  float    v8f;

union AV { v16h v; _Float16 h[16]; };
union BV { v16h v; uint4 q[2]; };

// ---------------------------------------------------------------------------
// Kernel 1: simulate the weight-dependent block on all 256 basis columns
// (one wave32 per column).  Lane holds 8 complex amps: i = (lane<<3)|r.
// Output: packed f16 B operands in WMMA register order, slab-major:
//   word index = ((mat*4 + iblk)*16 + kt*2+half)*512 + col*8 + v
//   (mat: 0=re 1=im; iblk = i>>6; col = i&63; kk=k>>1 -> kt=kk>>4,
//    half=(kk>>3)&1, v=kk&7; halfword parity = k&1)
// ---------------------------------------------------------------------------
__global__ void __launch_bounds__(128) vqc_columns_kernel(
    const float* __restrict__ wts,   // [4][8][2]
    void* __restrict__ ws) {
  const int lane = threadIdx.x & 31;
  const int wv   = threadIdx.x >> 5;
  const int k    = blockIdx.x * 4 + wv;       // column 0..255

  float re[8], im[8];
#pragma unroll
  for (int r = 0; r < 8; ++r) {
    int i = (lane << 3) | r;
    re[r] = (i == k) ? 1.f : 0.f;
    im[r] = 0.f;
  }

  for (int l = 0; l < 4; ++l) {
    // ---- CNOT ring: CNOT(q, (q+1)%8); qubit w lives at bit (7-w) ----
#pragma unroll
    for (int q = 0; q < 8; ++q) {
      const int t  = (q + 1) & 7;
      const int bc = 7 - q;
      const int bt = 7 - t;
      if (bt >= 3) {                       // target bit is a lane bit
        const int m = 1 << (bt - 3);
#pragma unroll
        for (int r = 0; r < 8; ++r) {
          int i = (lane << 3) | r;
          bool ctrl = (i >> bc) & 1;
          float pr = __shfl_xor(re[r], m, 32);
          float pi = __shfl_xor(im[r], m, 32);
          re[r] = ctrl ? pr : re[r];
          im[r] = ctrl ? pi : im[r];
        }
      } else {                             // target bit is a register bit
        const int mb = 1 << bt;
#pragma unroll
        for (int r = 0; r < 8; ++r) {
          if (!(r & mb)) {
            int r1 = r | mb;
            int i = (lane << 3) | r;       // ctrl bit identical for r and r1
            bool ctrl = (i >> bc) & 1;
            float t0 = re[r], t1 = re[r1];
            re[r]  = ctrl ? t1 : t0;  re[r1] = ctrl ? t0 : t1;
            t0 = im[r]; t1 = im[r1];
            im[r]  = ctrl ? t1 : t0;  im[r1] = ctrl ? t0 : t1;
          }
        }
      }
    }
    // ---- RY / RZ rotations ----
#pragma unroll
    for (int q = 0; q < 8; ++q) {
      const int b = 7 - q;
      float thy = wts[(l * 8 + q) * 2 + 0];
      float thz = wts[(l * 8 + q) * 2 + 1];
      float cy = cosf(0.5f * thy), sy = sinf(0.5f * thy);
      if (b >= 3) {
        const int m = 1 << (b - 3);
        float sg = ((lane >> (b - 3)) & 1) ? sy : -sy;
#pragma unroll
        for (int r = 0; r < 8; ++r) {
          float pr = __shfl_xor(re[r], m, 32);
          float pi = __shfl_xor(im[r], m, 32);
          re[r] = cy * re[r] + sg * pr;
          im[r] = cy * im[r] + sg * pi;
        }
      } else {
        const int mb = 1 << b;
#pragma unroll
        for (int r = 0; r < 8; ++r) {
          if (!(r & mb)) {
            int r1 = r | mb;
            float a0 = re[r], a1 = re[r1];
            re[r] = cy * a0 - sy * a1;  re[r1] = sy * a0 + cy * a1;
            a0 = im[r]; a1 = im[r1];
            im[r] = cy * a0 - sy * a1;  im[r1] = sy * a0 + cy * a1;
          }
        }
      }
      // RZ: diag(exp(-i thz/2), exp(+i thz/2)) on bit b
      float cz = cosf(0.5f * thz), sz = sinf(0.5f * thz);
#pragma unroll
      for (int r = 0; r < 8; ++r) {
        int i = (lane << 3) | r;
        float ci = ((i >> b) & 1) ? sz : -sz;
        float nr = re[r] * cz - im[r] * ci;
        float ni = re[r] * ci + im[r] * cz;
        re[r] = nr; im[r] = ni;
      }
    }
  }

  // ---- write swizzled packed output ----
  const int kk     = k >> 1;
  const int kt2h   = ((kk >> 4) << 1) | ((kk >> 3) & 1);  // kt*2 + half
  const int vc     = kk & 7;
  const int parity = k & 1;
  _Float16* g = (_Float16*)ws;
#pragma unroll
  for (int r = 0; r < 8; ++r) {
    int i    = (lane << 3) | r;
    int iblk = i >> 6;
    int col  = i & 63;
    int base = kt2h * 512 + col * 8 + vc;   // word within 8192-word slab
    g[(((0 * 4 + iblk) * 8192) + base) * 2 + parity] = (_Float16)re[r];
    g[(((1 * 4 + iblk) * 8192) + base) * 2 + parity] = (_Float16)im[r];
  }
}

// ---------------------------------------------------------------------------
// Kernel 2: out_b = sum_i z_i (|(V Cre^T)_{b,i}|^2 + |(V Cim^T)_{b,i}|^2)
// One wave per 16 batch rows.  A-tiles built in-register from x via factored
// product tree; B slabs (64 i-cols x K=256, re+im = 64KB) staged in LDS with
// linear uint4 copies; B tiles double-buffered (2x ds_load_b128 per operand).
// ---------------------------------------------------------------------------
__global__ void __launch_bounds__(128) vqc_eval_kernel(
    const float* __restrict__ x,          // [4096][8]
    const uint4* __restrict__ bpk4,       // 16384 uint4 (re: slabs 0..3, im: 4..7)
    float* __restrict__ out) {
  __shared__ uint4 lds4[4096];            // 64KB: re [0,2048), im [2048,4096)
  const int tid  = threadIdx.x;
  const int lane = tid & 31;
  const int wv   = tid >> 5;
  const int half = lane >> 4;
  const int n    = lane & 15;
  const int rowbase = (blockIdx.x * 4 + wv) * 16;
  const int row  = rowbase + n;

  // cos/sin of x/2 for this lane's batch row; term for index-bit b is
  // (bit ? sn[7-b] : cs[7-b])
  float cs[8], sn[8];
#pragma unroll
  for (int w = 0; w < 8; ++w) {
    float t = 0.5f * x[row * 8 + w];
    cs[w] = cosf(t);
    sn[w] = sinf(t);
  }

  // Factored product tree.  Lane's k values: bit4 == half (fixed),
  // bits {0,1,2,3} free (16 low products), bits {5,6,7} = kt (8 high products).
  float p01[4], p3l[8], plow[16], phigh[8];
#pragma unroll
  for (int m = 0; m < 4; ++m)
    p01[m] = ((m & 1) ? sn[7] : cs[7]) * ((m & 2) ? sn[6] : cs[6]);
#pragma unroll
  for (int m = 0; m < 8; ++m)
    p3l[m] = p01[m & 3] * ((m & 4) ? sn[5] : cs[5]);
  {
    float hterm = half ? sn[3] : cs[3];          // bit4 term (fixed per lane)
    float g0 = hterm * cs[4], g1 = hterm * sn[4]; // x bit3 term
#pragma unroll
    for (int m = 0; m < 8; ++m) { plow[m] = p3l[m] * g0; plow[8 + m] = p3l[m] * g1; }
  }
#pragma unroll
  for (int m = 0; m < 4; ++m)
    p01[m] = ((m & 1) ? sn[2] : cs[2]) * ((m & 2) ? sn[1] : cs[1]);
#pragma unroll
  for (int kt = 0; kt < 8; ++kt)
    phigh[kt] = p01[kt & 3] * ((kt & 4) ? sn[0] : cs[0]);

  // A-tiles (16x32 f16 layout: M=lane%16, K pair base = 2*v+16*half)
  AV at[8];
#pragma unroll
  for (int kt = 0; kt < 8; ++kt) {
#pragma unroll
    for (int v = 0; v < 8; ++v) {
#pragma unroll
      for (int p = 0; p < 2; ++p) {
        int lowidx = ((v >> 2) << 3) | ((v & 3) << 1) | p;  // bit3*8 + bits2..0
        at[kt].h[2 * v + p] = (_Float16)(phigh[kt] * plow[lowidx]);
      }
    }
  }

  float psum[8];
#pragma unroll
  for (int r = 0; r < 8; ++r) psum[r] = 0.f;

  for (int iblk = 0; iblk < 4; ++iblk) {
    __syncthreads();
    // stage slab: linear copy, fully coalesced (2048 uint4 per matrix)
#pragma unroll
    for (int j = 0; j < 16; ++j) {
      int w = tid + j * 128;
      lds4[w]        = bpk4[iblk * 2048 + w];            // re slab
      lds4[2048 + w] = bpk4[(4 + iblk) * 2048 + w];      // im slab
    }
    __syncthreads();

#pragma unroll
    for (int it = 0; it < 4; ++it) {
      int itile = iblk * 4 + it;
      v8f dre = {0.f, 0.f, 0.f, 0.f, 0.f, 0.f, 0.f, 0.f};
      v8f dim = {0.f, 0.f, 0.f, 0.f, 0.f, 0.f, 0.f, 0.f};
      BV br[2], bi[2];
      const int col2 = (it * 16 + n) * 2;
      {
        int idx = half * 128 + col2;                     // kt = 0
        br[0].q[0] = lds4[idx];        br[0].q[1] = lds4[idx + 1];
        bi[0].q[0] = lds4[2048 + idx]; bi[0].q[1] = lds4[2048 + idx + 1];
      }
#pragma unroll
      for (int kt = 0; kt < 8; ++kt) {
        const int cur = kt & 1, nxt = cur ^ 1;
        if (kt < 7) {                                    // prefetch next k-step
          int idx = ((kt + 1) * 2 + half) * 128 + col2;
          br[nxt].q[0] = lds4[idx];        br[nxt].q[1] = lds4[idx + 1];
          bi[nxt].q[0] = lds4[2048 + idx]; bi[nxt].q[1] = lds4[2048 + idx + 1];
        }
        dre = __builtin_amdgcn_wmma_f32_16x16x32_f16(
            false, at[kt].v, false, br[cur].v, (short)0, dre, false, false);
        dim = __builtin_amdgcn_wmma_f32_16x16x32_f16(
            false, at[kt].v, false, bi[cur].v, (short)0, dim, false, false);
        __builtin_amdgcn_sched_barrier(0);   // cap live B tiles at 2 k-steps
      }
      float z = (itile < 8) ? 1.f : -1.f;   // z_i by bit7 of i (itile>=8)
#pragma unroll
      for (int r = 0; r < 8; ++r)
        psum[r] += z * (dre[r] * dre[r] + dim[r] * dim[r]);
    }
  }

  // reduce over the 16 N-lanes (xor<16 stays within each half)
#pragma unroll
  for (int r = 0; r < 8; ++r) {
    float v = psum[r];
    v += __shfl_xor(v, 1, 32);
    v += __shfl_xor(v, 2, 32);
    v += __shfl_xor(v, 4, 32);
    v += __shfl_xor(v, 8, 32);
    psum[r] = v;
  }
  if (n == 0) {
#pragma unroll
    for (int r = 0; r < 8; ++r)
      out[rowbase + r + 8 * half] = psum[r];   // D row M = r + 8*half
  }
}

extern "C" void kernel_launch(void* const* d_in, const int* in_sizes, int n_in,
                              void* d_out, int out_size, void* d_ws, size_t ws_size,
                              hipStream_t stream) {
  (void)in_sizes; (void)n_in; (void)out_size; (void)ws_size;
  const float* x   = (const float*)d_in[0];     // [4096][8]
  const float* wts = (const float*)d_in[1];     // [4][8][2]
  float* out = (float*)d_out;                   // [4096]
  // workspace: 256KB packed f16 B operands (8 slabs x 8192 dwords)
  vqc_columns_kernel<<<64, 128, 0, stream>>>(wts, d_ws);
  vqc_eval_kernel<<<64, 128, 0, stream>>>(x, (const uint4*)d_ws, out);
}
